// ResidualAttentionBlock_65506841198862
// MI455X (gfx1250) — matrix operand order repaired and checked by reading, not compiled, not run on hardware
//
#include <hip/hip_runtime.h>
#include <math.h>

typedef __bf16 bf16;
typedef __attribute__((ext_vector_type(8)))  __bf16   v8bf;
typedef __attribute__((ext_vector_type(16))) __bf16   v16bf;
typedef __attribute__((ext_vector_type(8)))  float    v8f;
typedef __attribute__((ext_vector_type(4)))  unsigned u32x4;
typedef __attribute__((ext_vector_type(8)))  unsigned u32x8;

#define B_  2
#define T_  2048
#define D_  1024
#define H_  16
#define DH_ 64
#define M_  (B_*T_)

// ---------- helpers ----------

__device__ __forceinline__ float hash_noise(unsigned idx, unsigned seed) {
  unsigned x = idx ^ seed;
  x ^= x >> 16; x *= 0x7feb352dU;
  x ^= x >> 15; x *= 0x846ca68bU;
  x ^= x >> 16;
  return ((float)x * (1.0f / 4294967296.0f) - 0.5f) * 3.4641016f;
}

__device__ __forceinline__ v16bf cat8(v8bf lo, v8bf hi) {
  v16bf r;
#pragma unroll
  for (int i = 0; i < 8; ++i) { r[i] = lo[i]; r[8 + i] = hi[i]; }
  return r;
}

// A-fragment 16x32 (M x K) from a row-major [rows][ldk] bf16 LDS tile.
__device__ __forceinline__ v16bf frag_a(const bf16* tile, int row_base, int kbase,
                                        int ldk, int lane) {
  const int r  = row_base + (lane & 15);
  const int hw = lane >> 4;
  const bf16* p = tile + r * ldk + kbase + hw * 8;
  v8bf lo = *(const v8bf*)p;
  v8bf hi = *(const v8bf*)(p + 16);
  return cat8(lo, hi);
}

// B-fragment 32x16 (K x N) from an [N rows][K] row-major bf16 LDS tile.
__device__ __forceinline__ v16bf frag_b(const bf16* tile, int n_base, int kbase,
                                        int ldk, int lane) {
  const int n  = n_base + (lane & 15);
  const int hw = lane >> 4;
  const bf16* p = tile + n * ldk + kbase + hw * 8;
  v8bf lo = *(const v8bf*)p;
  v8bf hi = *(const v8bf*)(p + 16);
  return cat8(lo, hi);
}

// ---------- Tensor Data Mover: 2D strided bf16 tile -> padded LDS rows ----------
// D# per CDNA5 ISA. pad_interval/pad_amount are the
// encoded codes: pad after 8*2^interval bytes, pad (amount+1) DWORDs.
__device__ __forceinline__ void tdm_load_2d_bf16(
    unsigned lds_byte_addr, const bf16* gptr,
    unsigned tensor_d0, unsigned tensor_d1,
    unsigned tile_d0, unsigned tile_d1,
    unsigned stride_d0_elems,
    unsigned pad_interval, unsigned pad_amount) {
  unsigned long long ga = (unsigned long long)(size_t)gptr;
  u32x4 g0;
  g0[0] = 1u;                                   // count=1, user mode
  g0[1] = lds_byte_addr;                        // lds_addr
  g0[2] = (unsigned)(ga & 0xffffffffu);         // global_addr[31:0]
  g0[3] = (unsigned)((ga >> 32) & 0x1ffffffu)   // global_addr[56:32]
        | (2u << 30);                           // type=2 (image)
  u32x8 g1;
  g1[0] = (1u << 16)                            // data_size = 2 bytes
        | (1u << 20)                            // pad_enable
        | (pad_interval << 22)
        | (pad_amount << 25);
  g1[1] = (tensor_d0 & 0xffffu) << 16;          // tensor_dim0[15:0]
  g1[2] = (tensor_d0 >> 16) | ((tensor_d1 & 0xffffu) << 16);
  g1[3] = (tensor_d1 >> 16) | (tile_d0 << 16);  // tile_dim0
  g1[4] = tile_d1;                              // tile_dim1 (tile_dim2 = 0)
  g1[5] = stride_d0_elems;                      // tensor_dim0_stride[31:0]
  g1[6] = 0u;
  g1[7] = 0u;
  u32x4 gz = {0u, 0u, 0u, 0u};                  // groups 2/3: 2D tensor
  asm volatile("tensor_load_to_lds %0, %1, %2, %3"
               :: "s"(g0), "s"(g1), "s"(gz), "s"(gz)
               : "memory");
}

// ---------- fp32 -> bf16 pack ----------
__global__ __launch_bounds__(256) void cvt_f32_bf16(const float* __restrict__ s,
                                                    bf16* __restrict__ d, int n) {
  const int i = (blockIdx.x * blockDim.x + threadIdx.x) * 8;
  if (i < n) {
    float4 a = *(const float4*)(s + i);
    float4 b = *(const float4*)(s + i + 4);
    v8bf o = {(bf16)a.x, (bf16)a.y, (bf16)a.z, (bf16)a.w,
              (bf16)b.x, (bf16)b.y, (bf16)b.z, (bf16)b.w};
    *(v8bf*)(d + i) = o;
  }
}

// ---------- bf16 WMMA GEMM with TDM-staged tiles:  C = A[M,K] * W[N,K]^T ----------
// EPI: 0 = plain store, 1 = +bias +noise -> QuickGELU, 2 = +bias +residual(fp32)
template <int EPI, typename OutT>
__global__ __launch_bounds__(256) void gemm_bf16_wmma(
    const bf16* __restrict__ A, const bf16* __restrict__ W,
    const float* __restrict__ bias, const float* __restrict__ res,
    OutT* __restrict__ C, int Mdim, int Ndim, int Kdim,
    const float* __restrict__ stdp, unsigned seed) {
  constexpr int LDT = 40;  // 32 + 8 pad elems (80B rows) via TDM pad fields
  __shared__ __align__(16) bf16 As[2][128 * LDT];
  __shared__ __align__(16) bf16 Bs[2][128 * LDT];

  const int tid  = threadIdx.x;
  const int lane = tid & 31;
  const int wid  = tid >> 5;
  const int wm   = wid & 3;   // 4 waves over M (32 rows each)
  const int wn   = wid >> 2;  // 2 waves over N (64 cols each)
  const int m0   = blockIdx.y * 128;
  const int n0   = blockIdx.x * 128;
  const bool issuer = (wid == 0);  // wave-uniform

  v8f acc[2][4];
#pragma unroll
  for (int i = 0; i < 2; ++i)
#pragma unroll
    for (int j = 0; j < 4; ++j)
#pragma unroll
      for (int e = 0; e < 8; ++e) acc[i][j][e] = 0.0f;

  const int ksteps = Kdim >> 5;
  if (issuer) {  // prologue: tile 0 into buffer 0 (pad: 64B interval, 16B pad)
    tdm_load_2d_bf16((unsigned)(size_t)&As[0][0], A + (size_t)m0 * Kdim,
                     (unsigned)Kdim, (unsigned)Mdim, 32u, 128u, (unsigned)Kdim, 3u, 3u);
    tdm_load_2d_bf16((unsigned)(size_t)&Bs[0][0], W + (size_t)n0 * Kdim,
                     (unsigned)Kdim, (unsigned)Ndim, 32u, 128u, (unsigned)Kdim, 3u, 3u);
  }

  for (int kt = 0; kt < ksteps; ++kt) {
    if (issuer) {
      if (kt + 1 < ksteps) {  // keep next tile in flight, wait for current
        const int nb = (kt + 1) & 1;
        const int k1 = (kt + 1) << 5;
        tdm_load_2d_bf16((unsigned)(size_t)&As[nb][0], A + (size_t)m0 * Kdim + k1,
                         (unsigned)Kdim, (unsigned)Mdim, 32u, 128u, (unsigned)Kdim, 3u, 3u);
        tdm_load_2d_bf16((unsigned)(size_t)&Bs[nb][0], W + (size_t)n0 * Kdim + k1,
                         (unsigned)Kdim, (unsigned)Ndim, 32u, 128u, (unsigned)Kdim, 3u, 3u);
        __builtin_amdgcn_s_wait_tensorcnt(2);
      } else {
        __builtin_amdgcn_s_wait_tensorcnt(0);
      }
    }
    __syncthreads();

    const bf16* at = &As[kt & 1][0];
    const bf16* bt = &Bs[kt & 1][0];
    v16bf af[2], bfr[4];
#pragma unroll
    for (int mf = 0; mf < 2; ++mf) af[mf] = frag_a(at, wm * 32 + mf * 16, 0, LDT, lane);
#pragma unroll
    for (int nf = 0; nf < 4; ++nf) bfr[nf] = frag_b(bt, wn * 64 + nf * 16, 0, LDT, lane);
#pragma unroll
    for (int mf = 0; mf < 2; ++mf)
#pragma unroll
      for (int nf = 0; nf < 4; ++nf)
        acc[mf][nf] = __builtin_amdgcn_wmma_f32_16x16x32_bf16(
            false, af[mf], false, bfr[nf], (short)0, acc[mf][nf], false, false);
    __syncthreads();  // buffer consumed; safe to overwrite next iteration
  }

  const float stdv = stdp[0];
  const int hw = lane >> 4, l15 = lane & 15;
#pragma unroll
  for (int mf = 0; mf < 2; ++mf)
#pragma unroll
    for (int nf = 0; nf < 4; ++nf)
#pragma unroll
      for (int j = 0; j < 8; ++j) {
        const int row = m0 + wm * 32 + mf * 16 + j + hw * 8;
        const int col = n0 + wn * 64 + nf * 16 + l15;
        const size_t idx = (size_t)row * Ndim + col;
        float v = acc[mf][nf][j];
        if (EPI == 1) {
          v += bias[col];
          v += stdv * hash_noise((unsigned)idx, seed);
          v = v / (1.0f + __expf(-1.702f * v));  // QuickGELU
        } else if (EPI == 2) {
          v += bias[col] + res[idx];
        }
        C[idx] = (OutT)v;
      }
}

// ---------- flash attention over bf16 qkv[B,T,H,3*DH]; writes b = softmax(qk/s)v ----------
__global__ __launch_bounds__(256) void attn_flash_wmma(
    const bf16* __restrict__ qkv, float* __restrict__ bout,
    const float* __restrict__ stdp, unsigned seed) {
  constexpr int LQ = 72;  // 64 + 8 pad elems (144B rows) via TDM pad fields
  __shared__ __align__(16) bf16 Qs[128 * LQ];
  __shared__ __align__(16) bf16 Ks[64 * LQ];
  __shared__ __align__(16) bf16 Vt[64 * LQ];   // V transposed: [DH][S-chunk]
  __shared__ __align__(16) bf16 Ps[128 * LQ];

  const int tid  = threadIdx.x;
  const int lane = tid & 31;
  const int wid  = tid >> 5;
  const int bh   = blockIdx.x & (B_ * H_ - 1);
  const int mt   = blockIdx.x >> 5;
  const int b    = bh >> 4;
  const int h    = bh & (H_ - 1);
  const int m0   = mt * 128;
  const float stdv = stdp[0];
  const bool issuer = (wid == 0);
  const unsigned row_stride = H_ * 3 * DH_;  // 192*H elems between t rows

  if (issuer) {  // Q tile: 128 x 64, strided rows (pad: 128B interval, 16B pad)
    tdm_load_2d_bf16((unsigned)(size_t)&Qs[0],
                     qkv + ((size_t)(b * T_ + m0) * H_ + h) * 192,
                     64u, 128u, 64u, 128u, row_stride, 4u, 3u);
  }

  float mrow[8], lrow[8];
  v8f oacc[4];
#pragma unroll
  for (int j = 0; j < 8; ++j) { mrow[j] = -1e30f; lrow[j] = 0.0f; }
#pragma unroll
  for (int nf = 0; nf < 4; ++nf)
#pragma unroll
    for (int e = 0; e < 8; ++e) oacc[nf][e] = 0.0f;

  const int hw = lane >> 4, l15 = lane & 15;

  for (int s0 = 0; s0 < T_; s0 += 64) {
    __syncthreads();  // previous chunk fully consumed
    if (issuer) {     // K chunk: 64 x 64 strided tile via TDM
      tdm_load_2d_bf16((unsigned)(size_t)&Ks[0],
                       qkv + ((size_t)(b * T_ + s0) * H_ + h) * 192 + DH_,
                       64u, 64u, 64u, 64u, row_stride, 4u, 3u);
    }
    {  // V chunk staged manually with transpose (TDM cannot transpose)
      const int r  = tid >> 2;
      const int q4 = tid & 3;
      const bf16* base = qkv + ((size_t)(b * T_ + s0 + r) * H_ + h) * 192 + 2 * DH_;
      v8bf v0 = *(const v8bf*)(base + q4 * 16);
      v8bf v1 = *(const v8bf*)(base + q4 * 16 + 8);
#pragma unroll
      for (int i = 0; i < 8; ++i) Vt[(q4 * 16 + i) * LQ + r]     = v0[i];
#pragma unroll
      for (int i = 0; i < 8; ++i) Vt[(q4 * 16 + 8 + i) * LQ + r] = v1[i];
      if (s0 + 64 < T_)  // global_prefetch_b8 next chunk
        __builtin_prefetch(base + (size_t)64 * row_stride, 0, 1);
    }
    if (issuer) __builtin_amdgcn_s_wait_tensorcnt(0);
    __syncthreads();

    // S = Q * K^T : 16 rows x 64 cols per wave
    v8f sacc[4];
#pragma unroll
    for (int nf = 0; nf < 4; ++nf)
#pragma unroll
      for (int e = 0; e < 8; ++e) sacc[nf][e] = 0.0f;
#pragma unroll
    for (int kb = 0; kb < DH_; kb += 32) {
      v16bf aq = frag_a(Qs, wid * 16, kb, LQ, lane);
#pragma unroll
      for (int nf = 0; nf < 4; ++nf) {
        v16bf bk = frag_b(Ks, nf * 16, kb, LQ, lane);
        sacc[nf] = __builtin_amdgcn_wmma_f32_16x16x32_bf16(
            false, aq, false, bk, (short)0, sacc[nf], false, false);
      }
    }

    // online softmax; row j+8*hw lives in one half-wave -> 16-lane reductions
#pragma unroll
    for (int j = 0; j < 8; ++j) {
      const int tg = m0 + wid * 16 + j + hw * 8;
      float tv[4];
      float mx = mrow[j];
#pragma unroll
      for (int nf = 0; nf < 4; ++nf) {
        const int sg = s0 + nf * 16 + l15;
        float v = sacc[nf][j] * 0.125f;  // 1/sqrt(DH) folded here
        v += stdv * hash_noise((unsigned)(((b * T_ + tg) * T_ + sg) * H_ + h), seed);
        tv[nf] = v;
        mx = fmaxf(mx, v);
      }
#pragma unroll
      for (int off = 1; off < 16; off <<= 1) mx = fmaxf(mx, __shfl_xor(mx, off, 32));
      const float alpha = __expf(mrow[j] - mx);
      mrow[j] = mx;
      float sum = 0.0f;
#pragma unroll
      for (int nf = 0; nf < 4; ++nf) {
        const float p = __expf(tv[nf] - mx);
        sum += p;
        Ps[(wid * 16 + j + hw * 8) * LQ + nf * 16 + l15] = (bf16)p;
      }
#pragma unroll
      for (int off = 1; off < 16; off <<= 1) sum += __shfl_xor(sum, off, 32);
      lrow[j] = lrow[j] * alpha + sum;
#pragma unroll
      for (int nf = 0; nf < 4; ++nf) oacc[nf][j] *= alpha;
    }

    // O += P * V  (Ps rows per-wave exclusive; same-wave DS ops are in-order)
#pragma unroll
    for (int kb = 0; kb < 64; kb += 32) {
      v16bf ap = frag_a(Ps, wid * 16, kb, LQ, lane);
#pragma unroll
      for (int nf = 0; nf < 4; ++nf) {
        v16bf bv = frag_b(Vt, nf * 16, kb, LQ, lane);
        oacc[nf] = __builtin_amdgcn_wmma_f32_16x16x32_bf16(
            false, ap, false, bv, (short)0, oacc[nf], false, false);
      }
    }
  }

  // write b (pre-residual attention output), layout [B,T,D], fp32
#pragma unroll
  for (int j = 0; j < 8; ++j) {
    const int tg = m0 + wid * 16 + j + hw * 8;
    const float inv = 1.0f / lrow[j];
#pragma unroll
    for (int nf = 0; nf < 4; ++nf) {
      const int dh = nf * 16 + l15;
      bout[(size_t)(b * T_ + tg) * D_ + h * DH_ + dh] = oacc[nf][j] * inv;
    }
  }
}

// ---------- x2 = 2*x + b (fp32 + bf16 copies) ----------
__global__ __launch_bounds__(256) void residual2_kernel(
    const float* __restrict__ x, const float* __restrict__ battn,
    float* __restrict__ x2f, bf16* __restrict__ x2b, int n) {
  const int i = blockIdx.x * blockDim.x + threadIdx.x;
  if (i < n) {
    const float v = 2.0f * x[i] + battn[i];
    x2f[i] = v;
    x2b[i] = (bf16)v;
  }
}

extern "C" void kernel_launch(void* const* d_in, const int* in_sizes, int n_in,
                              void* d_out, int out_size, void* d_ws, size_t ws_size,
                              hipStream_t stream) {
  const float* x    = (const float*)d_in[0];
  const float* Wqkv = (const float*)d_in[1];
  const float* Wfo  = (const float*)d_in[2];
  const float* bfo  = (const float*)d_in[3];
  const float* Wfi  = (const float*)d_in[4];
  const float* bfi  = (const float*)d_in[5];
  const float* stdp = (const float*)d_in[6];
  float* out = (float*)d_out;

  char* w = (char*)d_ws;
  bf16*  xb    = (bf16*)w;  w += (size_t)M_ * D_ * 2;        // x in bf16
  bf16*  Wqkvb = (bf16*)w;  w += (size_t)3 * D_ * D_ * 2;
  bf16*  Wfob  = (bf16*)w;  w += (size_t)3 * D_ * D_ * 2;
  bf16*  Wfib  = (bf16*)w;  w += (size_t)3 * D_ * D_ * 2;
  bf16*  qkvb  = (bf16*)w;  w += (size_t)M_ * 3 * D_ * 2;    // qkv bf16
  float* battn = (float*)w; w += (size_t)M_ * D_ * 4;        // attention b
  float* x2f   = (float*)w; w += (size_t)M_ * D_ * 4;        // 2x+b fp32
  bf16*  x2b   = (bf16*)w;  w += (size_t)M_ * D_ * 2;        // 2x+b bf16
  bf16*  hb    = (bf16*)w;                                   // gelu(mlp up) bf16

  const int nW = 3 * D_ * D_;
  const int nX = M_ * D_;
  // 0) one-time fp32 -> bf16 conversions (amortized across all K-reuse)
  cvt_f32_bf16<<<(nX / 8 + 255) / 256, 256, 0, stream>>>(x, xb, nX);
  cvt_f32_bf16<<<(nW / 8 + 255) / 256, 256, 0, stream>>>(Wqkv, Wqkvb, nW);
  cvt_f32_bf16<<<(nW / 8 + 255) / 256, 256, 0, stream>>>(Wfo, Wfob, nW);
  cvt_f32_bf16<<<(nW / 8 + 255) / 256, 256, 0, stream>>>(Wfi, Wfib, nW);

  // 1) qkv = x @ Wqkv^T  (bf16 out)
  gemm_bf16_wmma<0, bf16><<<dim3(3 * D_ / 128, M_ / 128), 256, 0, stream>>>(
      xb, Wqkvb, nullptr, nullptr, qkvb, M_, 3 * D_, D_, stdp, 0u);
  // 2) b = attention(qkv) with score noise
  attn_flash_wmma<<<dim3((T_ / 128) * B_ * H_), 256, 0, stream>>>(
      qkvb, battn, stdp, 0x9E3779B9u);
  // 3) x2 = 2x + b (double residual)
  residual2_kernel<<<(nX + 255) / 256, 256, 0, stream>>>(x, battn, x2f, x2b, nX);
  // 4) h = quick_gelu(x2 @ Wfo^T + bfo + noise)  (bf16 out)
  gemm_bf16_wmma<1, bf16><<<dim3(3 * D_ / 128, M_ / 128), 256, 0, stream>>>(
      x2b, Wfob, bfo, nullptr, hb, M_, 3 * D_, D_, stdp, 0x85EBCA6Bu);
  // 5) out = x2 + h @ Wfi^T + bfi  (fp32 out)
  gemm_bf16_wmma<2, float><<<dim3(D_ / 128, M_ / 128), 256, 0, stream>>>(
      hb, Wfib, bfi, x2f, out, M_, D_, 3 * D_, stdp, 0u);
}
